// VectorQuantizer_27625229648508
// MI455X (gfx1250) — matrix operand order repaired and checked by reading, not compiled, not run on hardware
//
#include <hip/hip_runtime.h>
#include <hip/hip_bf16.h>

// ---------------------------------------------------------------------------
// VQ-VAE vector quantizer for MI455X (gfx1250, wave32, WMMA).
// Distances via bf16 split-GEMM (hi/lo) on v_wmma_f32_16x16x32_bf16,
// fp32 accumulate; argmin drops the per-row ||x||^2 constant.
// B tiles staged with CDNA5 async global->LDS DMA (ASYNCcnt), double-buffered.
// ---------------------------------------------------------------------------

typedef __attribute__((ext_vector_type(16))) __bf16          v16bf;
typedef __attribute__((ext_vector_type(8)))  float           v8f;
typedef __attribute__((ext_vector_type(16))) unsigned short  v16u;

#define NROWS   32768     // B*T
#define DDIM    256
#define KCB     1024
#define NTILES  64        // KCB / 16
#define KCHUNKS 8         // DDIM / 32
#define TILE_BYTES 16384  // one staged B tile (hi+lo, 8 chunks)

// workspace layout (bytes)
#define WS_PACKED 0                         // 1 MB: packed bf16 codebook (hi/lo fragments)
#define WS_NORMS  (1048576)                 // 4 KB: ||e_k||^2
#define WS_COUNTS (1048576 + 4096)          // 4 KB: usage counts
#define WS_SSE    (1048576 + 8192)          // 4 B : sum of squared error

// output layout (float elements)
#define OUT_QUANT 0
#define OUT_LOSS  8388608
#define OUT_PPL   8388609
#define OUT_IDX   8388610

__device__ __forceinline__ unsigned short f2bf(float f) {
  unsigned int u = __float_as_uint(f);
  u += 0x7FFFu + ((u >> 16) & 1u);          // round-to-nearest-even
  return (unsigned short)(u >> 16);
}
__device__ __forceinline__ float bf2f(unsigned short h) {
  return __uint_as_float(((unsigned int)h) << 16);
}

// Async DMA of 64 contiguous bytes per lane: global -> LDS, no VGPR data path.
// GVS mode: addr = SADDR64 + VADDR32 + IOFFSET; IOFFSET is added to BOTH the
// global and the LDS address (ISA 08_async_tensor §4.4), so one base pair +
// offset:0/16/32/48 copies 64B. Tracked by ASYNCcnt.
__device__ __forceinline__ void async_copy64(unsigned lds_addr, unsigned goff,
                                             const void* sbase) {
  asm volatile(
      "global_load_async_to_lds_b128 %0, %1, %2 offset:0\n\t"
      "global_load_async_to_lds_b128 %0, %1, %2 offset:16\n\t"
      "global_load_async_to_lds_b128 %0, %1, %2 offset:32\n\t"
      "global_load_async_to_lds_b128 %0, %1, %2 offset:48"
      :
      : "v"(lds_addr), "v"(goff), "s"(sbase)
      : "memory");
}
__device__ __forceinline__ void wait_async0() {
  asm volatile("s_wait_asynccnt 0x0" ::: "memory");
}

// ---------------------------------------------------------------------------
// Kernel 1: pack codebook into WMMA B-fragment order (hi/lo bf16) + zero accums.
// B fragment (16-bit, 32x16): lane L holds N = L%16, koff = (L<16?0:8); K = koff+i (i<8), K = koff+8+i (i>=8).
// Packed element index: (((n*8 + c)*2 + buf)*32 + lane)*16 + i
// ---------------------------------------------------------------------------
__global__ void vq_pack(const float* __restrict__ cb,
                        unsigned short* __restrict__ packed,
                        unsigned int* __restrict__ counts,
                        float* __restrict__ gsse) {
  int tid = blockIdx.x * blockDim.x + threadIdx.x;   // 16384 threads
  if (tid < KCB) counts[tid] = 0u;
  if (tid == 0)  *gsse = 0.0f;

  int lane = tid & 31;
  int c    = (tid >> 5) & 7;
  int n    = tid >> 8;

  const float* src = cb + (size_t)(n * 16 + (lane & 15)) * DDIM
                        + c * 32 + ((lane < 16) ? 0 : 8);
  float4 a0 = *(const float4*)(src + 0);
  float4 a1 = *(const float4*)(src + 4);
  float4 a2 = *(const float4*)(src + 16);
  float4 a3 = *(const float4*)(src + 20);
  float f[16] = {a0.x, a0.y, a0.z, a0.w, a1.x, a1.y, a1.z, a1.w,
                 a2.x, a2.y, a2.z, a2.w, a3.x, a3.y, a3.z, a3.w};
  v16u hi, lo;
#pragma unroll
  for (int i = 0; i < 16; ++i) {
    unsigned short hb = f2bf(f[i]);
    hi[i] = hb;
    lo[i] = f2bf(f[i] - bf2f(hb));
  }
  size_t baseHi = ((size_t)((n * 8 + c) * 2 + 0) * 32 + lane) * 16;
  size_t baseLo = ((size_t)((n * 8 + c) * 2 + 1) * 32 + lane) * 16;
  *(v16u*)(packed + baseHi) = hi;
  *(v16u*)(packed + baseLo) = lo;
}

// ---------------------------------------------------------------------------
// Kernel 2: codebook squared norms (fp32 exact).
// ---------------------------------------------------------------------------
__global__ void vq_norms(const float* __restrict__ cb, float* __restrict__ norms) {
  int k = blockIdx.x * blockDim.x + threadIdx.x;
  if (k >= KCB) return;
  const float4* p = (const float4*)(cb + (size_t)k * DDIM);
  float s = 0.0f;
#pragma unroll 4
  for (int i = 0; i < DDIM / 4; ++i) {
    float4 v = p[i];
    s += v.x * v.x + v.y * v.y + v.z * v.z + v.w * v.w;
  }
  norms[k] = s;
}

// ---------------------------------------------------------------------------
// Kernel 3: main. 256 blocks x 256 threads (8 waves). Wave w owns 16 rows.
// ---------------------------------------------------------------------------
__launch_bounds__(256)
__global__ void vq_main(const float* __restrict__ X,
                        const float* __restrict__ cb,
                        const unsigned short* __restrict__ packed,
                        const float* __restrict__ norms,
                        unsigned int* __restrict__ counts,
                        float* __restrict__ gsse,
                        float* __restrict__ outQ,
                        float* __restrict__ outIdx) {
  __shared__ uint4 Btile[2048];        // 2 x 16 KB double-buffered B tile
  __shared__ float redV[8][16][16];    // per-wave argmin candidates
  __shared__ int   redI[8][16][16];
  __shared__ int   rowIdxS[8][16];
  __shared__ float normsS[KCB];        // 4 KB

  const int t    = threadIdx.x;
  const int lane = t & 31;
  const int w    = t >> 5;
  const int rowBase = blockIdx.x * 128 + w * 16;

  for (int i = t; i < KCB; i += 256) normsS[i] = norms[i];

  // Build A hi/lo fragments for this wave's 16 rows.
  // A fragment (16-bit, 16x32): lane L -> M = L%16; koff = (L<16?0:8);
  // element i: K = koff + i (i<8), K = koff + 8 + i (i>=8).
  const int m    = lane & 15;
  const int koff = (lane < 16) ? 0 : 8;
  v16bf ahi[KCHUNKS], alo[KCHUNKS];
  const float* xrow = X + (size_t)(rowBase + m) * DDIM + koff;
#pragma unroll
  for (int c = 0; c < KCHUNKS; ++c) {
    const float* p = xrow + c * 32;
    float4 a0 = *(const float4*)(p + 0);
    float4 a1 = *(const float4*)(p + 4);
    float4 a2 = *(const float4*)(p + 16);
    float4 a3 = *(const float4*)(p + 20);
    float f[16] = {a0.x, a0.y, a0.z, a0.w, a1.x, a1.y, a1.z, a1.w,
                   a2.x, a2.y, a2.z, a2.w, a3.x, a3.y, a3.z, a3.w};
    v16u h, l;
#pragma unroll
    for (int i = 0; i < 16; ++i) {
      unsigned short hb = f2bf(f[i]);
      h[i] = hb;
      l[i] = f2bf(f[i] - bf2f(hb));
    }
    ahi[c] = __builtin_bit_cast(v16bf, h);
    alo[c] = __builtin_bit_cast(v16bf, l);
  }

  float bestV[8];
  int   bestI[8];
#pragma unroll
  for (int j = 0; j < 8; ++j) { bestV[j] = 3.4e38f; bestI[j] = 0; }

  char* btile = (char*)Btile;
  const unsigned ldsBase = (unsigned)(size_t)btile;   // LDS offset of Btile

  // prologue: async-stage tile 0 into buffer 0 (each thread copies 64 B)
  async_copy64(ldsBase + (unsigned)(t * 64), (unsigned)(t * 64), packed);

  for (int n = 0; n < NTILES; ++n) {
    wait_async0();      // this wave's DMA slice done
    __syncthreads();    // every wave's slice done; prev buffer fully consumed

    if (n + 1 < NTILES) {
      unsigned nb = (unsigned)((n + 1) & 1) * TILE_BYTES;
      async_copy64(ldsBase + nb + (unsigned)(t * 64),
                   (unsigned)(n + 1) * TILE_BYTES + (unsigned)(t * 64), packed);
    }

    const char* bbuf = btile + (n & 1) * TILE_BYTES;
    v8f acc = {};
#pragma unroll
    for (int c = 0; c < KCHUNKS; ++c) {
      v16bf bh = *(const v16bf*)(bbuf + (((c * 2 + 0) * 32 + lane) * 32));
      v16bf bl = *(const v16bf*)(bbuf + (((c * 2 + 1) * 32 + lane) * 32));
      // split-precision fp32 emulation: x.e ~= xh.eh + xh.el + xl.eh
      acc = __builtin_amdgcn_wmma_f32_16x16x32_bf16(false, ahi[c], false, bh,
                                                    (short)0, acc, false, false);
      acc = __builtin_amdgcn_wmma_f32_16x16x32_bf16(false, ahi[c], false, bl,
                                                    (short)0, acc, false, false);
      acc = __builtin_amdgcn_wmma_f32_16x16x32_bf16(false, alo[c], false, bh,
                                                    (short)0, acc, false, false);
    }
    // C/D layout: lane -> N = lane%16; VGPR j -> M = (lane<16?0:8)+j
    int   kcol = n * 16 + m;
    float nrm  = normsS[kcol];
#pragma unroll
    for (int j = 0; j < 8; ++j) {
      float d = nrm - 2.0f * acc[j];   // ||x||^2 dropped (row-constant)
      if (d < bestV[j]) { bestV[j] = d; bestI[j] = kcol; }  // kcol ascending -> first-min kept
    }
  }

  // per-row argmin across 16 lanes
  {
    int mrow = (lane < 16) ? 0 : 8;
#pragma unroll
    for (int j = 0; j < 8; ++j) {
      redV[w][mrow + j][m] = bestV[j];
      redI[w][mrow + j][m] = bestI[j];
    }
  }
  __syncthreads();
  if (lane < 16) {
    float bv = 3.4e38f;
    int   bi = 0x7FFFFFFF;
#pragma unroll
    for (int col = 0; col < 16; ++col) {
      float v  = redV[w][lane][col];
      int   id = redI[w][lane][col];
      if (v < bv || (v == bv && id < bi)) { bv = v; bi = id; }
    }
    rowIdxS[w][lane] = bi;
    outIdx[rowBase + lane] = (float)bi;
    atomicAdd(&counts[bi], 1u);
  }
  __syncthreads();

  // gather codebook rows (fp32 exact) -> quantized output; accumulate SSE
  float sse = 0.0f;
#pragma unroll 4
  for (int r = 0; r < 16; ++r) {
    int k = rowIdxS[w][r];
    const float4* e4 = (const float4*)(cb + (size_t)k * DDIM);
    const float4* x4 = (const float4*)(X + (size_t)(rowBase + r) * DDIM);
    float4*       o4 = (float4*)(outQ + (size_t)(rowBase + r) * DDIM);
    int c = lane * 2;
    float4 e0 = e4[c], e1 = e4[c + 1];
    float4 v0 = x4[c], v1 = x4[c + 1];
    o4[c] = e0; o4[c + 1] = e1;
    float d;
    d = e0.x - v0.x; sse += d * d;
    d = e0.y - v0.y; sse += d * d;
    d = e0.z - v0.z; sse += d * d;
    d = e0.w - v0.w; sse += d * d;
    d = e1.x - v1.x; sse += d * d;
    d = e1.y - v1.y; sse += d * d;
    d = e1.z - v1.z; sse += d * d;
    d = e1.w - v1.w; sse += d * d;
  }
#pragma unroll
  for (int off = 16; off >= 1; off >>= 1) sse += __shfl_xor(sse, off, 32);
  if (lane == 0) atomicAdd(gsse, sse);
}

// ---------------------------------------------------------------------------
// Kernel 4: loss + perplexity.
// ---------------------------------------------------------------------------
__global__ void vq_finalize(const unsigned int* __restrict__ counts,
                            const float* __restrict__ gsse,
                            float* __restrict__ outLoss,
                            float* __restrict__ outPpl) {
  __shared__ float s[256];
  int t = threadIdx.x;
  float acc = 0.0f;
  for (int i = t; i < KCB; i += 256) {
    float p = (float)counts[i] * (1.0f / (float)NROWS);
    acc += p * logf(p + 1e-10f);
  }
  s[t] = acc;
  __syncthreads();
  for (int off = 128; off >= 1; off >>= 1) {
    if (t < off) s[t] += s[t + off];
    __syncthreads();
  }
  if (t == 0) {
    *outPpl  = expf(-s[0]);
    *outLoss = (*gsse) * 1.25f / (float)(NROWS * DDIM);   // q_latent + 0.25*e_latent
  }
}

// ---------------------------------------------------------------------------
extern "C" void kernel_launch(void* const* d_in, const int* in_sizes, int n_in,
                              void* d_out, int out_size, void* d_ws, size_t ws_size,
                              hipStream_t stream) {
  const float* X  = (const float*)d_in[0];   // [32,1024,256] fp32
  const float* cb = (const float*)d_in[1];   // [1024,256] fp32

  unsigned short* packed = (unsigned short*)((char*)d_ws + WS_PACKED);
  float*          norms  = (float*)((char*)d_ws + WS_NORMS);
  unsigned int*   counts = (unsigned int*)((char*)d_ws + WS_COUNTS);
  float*          gsse   = (float*)((char*)d_ws + WS_SSE);

  float* out     = (float*)d_out;
  float* outQ    = out + OUT_QUANT;
  float* outLoss = out + OUT_LOSS;
  float* outPpl  = out + OUT_PPL;
  float* outIdx  = out + OUT_IDX;

  vq_pack    <<<64,  256, 0, stream>>>(cb, packed, counts, gsse);
  vq_norms   <<<4,   256, 0, stream>>>(cb, norms);
  vq_main    <<<256, 256, 0, stream>>>(X, cb, packed, norms, counts, gsse, outQ, outIdx);
  vq_finalize<<<1,   256, 0, stream>>>(counts, gsse, outLoss, outPpl);
}
